// ForwardEulerSimulator_22196390985735
// MI455X (gfx1250) — matrix operand order repaired and checked by reading, not compiled, not run on hardware
//
#include <hip/hip_runtime.h>

typedef __attribute__((ext_vector_type(16))) _Float16 v16h;
typedef __attribute__((ext_vector_type(2)))  __fp16   fp16x2;
typedef __attribute__((ext_vector_type(8)))  float    v8f;

#define M_STEPS 1024
#define Q_BATCH 4096
#define NXS 10
#define NUS 4
#define NUIS 4
#define HID 64
#define CGM_MIN -0.8f
#define CGM_MAX 0.8f

__device__ __forceinline__ v8f wmma16x16x32(v16h a, v16h b, v8f c) {
  // (neg_a, A, neg_b, B, c_mod, C, reuse_a, reuse_b)
  return __builtin_amdgcn_wmma_f32_16x16x32_f16(false, a, false, b, (short)0, c, false, false);
}

__device__ __forceinline__ float tanh_fast(float x) {
#if __has_builtin(__builtin_amdgcn_tanhf)
  return __builtin_amdgcn_tanhf(x);   // v_tanh_f32 (gfx1250 TRANS op)
#else
  return tanhf(x);
#endif
}

// pack two f32 into one dword of two f16 (v_cvt_pk_rtz_f16_f32)
__device__ __forceinline__ unsigned pk(float a, float b) {
  union { fp16x2 h; unsigned u; } c;
  c.h = __builtin_amdgcn_cvt_pkrtz(a, b);
  return c.u;
}

union BU { v16h h; unsigned u[8]; };

// A-matrix 16x32 f16 layout (contiguous, matching B): lane<16: K=j
//                                                     lane>=16: K=16+j
__device__ __forceinline__ int kmapA(int j, int hi) {
  return j + (hi ? 16 : 0);
}

// One 16x32 A-tile of W1^T (hidden-block hb), bias folded into column K==Kin.
__device__ __forceinline__ v16h load_w1_tile(const float* __restrict__ W1,
                                             const float* __restrict__ b1,
                                             int Kin, int hb, int lane) {
  const int hi = lane >> 4;
  const int mh = hb * 16 + (lane & 15);
  v16h a;
#pragma unroll
  for (int j = 0; j < 16; ++j) {
    const int k = kmapA(j, hi);
    float v = 0.f;
    if (k < Kin)       v = W1[k * HID + mh];   // W1^T[mh][k]
    else if (k == Kin) v = b1[mh];             // bias column (B carries 1.0 at K==Kin)
    a[j] = (_Float16)v;
  }
  return a;
}

// One 16x32 A-tile of W2^T (K-block kb over hidden dim), rows >= NXS zeroed.
__device__ __forceinline__ v16h load_w2_tile(const float* __restrict__ W2, int kb, int lane) {
  const int hi = lane >> 4;
  const int m = lane & 15;
  v16h a;
#pragma unroll
  for (int j = 0; j < 16; ++j) {
    const int k = kb * 32 + kmapA(j, hi);
    const float v = (m < NXS) ? W2[k * NXS + m] : 0.f;  // W2^T[m][k]
    a[j] = (_Float16)v;
  }
  return a;
}

// Build one 32x16 f16 B-tile of h^T from two adjacent C-layout hidden tiles.
// B layout: lane<16: K=j; lane>=16: K=16+j. Lo half needs partner's P, hi half
// needs partner's Q -> select first, ONE shuffle per dword (4 per tile).
__device__ __forceinline__ v16h make_bh(const v8f& t0, const v8f& t1, int hi) {
  unsigned P[4], Q[4];
#pragma unroll
  for (int d = 0; d < 4; ++d) {
    P[d] = pk(t0[2 * d], t0[2 * d + 1]);
    Q[d] = pk(t1[2 * d], t1[2 * d + 1]);
  }
  BU o;
#pragma unroll
  for (int d = 0; d < 4; ++d) {
    const unsigned T  = hi ? P[d] : Q[d];               // value the partner needs
    const unsigned Ts = (unsigned)__shfl_xor((int)T, 16, 32);
    o.u[d]     = hi ? Ts   : P[d];
    o.u[d + 4] = hi ? Q[d] : Ts;
  }
  return o.h;
}

__global__ __launch_bounds__(32) void fes_kernel(
    const float* __restrict__ x0, const float* __restrict__ u_b, const float* __restrict__ ui_b,
    const float* __restrict__ W1p, const float* __restrict__ b1p,
    const float* __restrict__ W2p, const float* __restrict__ b2p,
    const float* __restrict__ W1i, const float* __restrict__ b1i,
    const float* __restrict__ W2i, const float* __restrict__ b2i,
    float* __restrict__ out) {
  const int lane = threadIdx.x & 31;
  const int hi   = lane >> 4;
  const int b    = blockIdx.x * 16 + (lane & 15);

  // ---- Weights resident in VGPRs for the whole rollout (A-layout, transposed GEMMs) ----
  v16h A1p[4], A1i[4];
#pragma unroll
  for (int hb = 0; hb < 4; ++hb) {
    A1p[hb] = load_w1_tile(W1p, b1p, NXS + NUS,        hb, lane);
    A1i[hb] = load_w1_tile(W1i, b1i, NXS + NUS + NUIS, hb, lane);
  }
  v16h A2p[2], A2i[2];
#pragma unroll
  for (int kb = 0; kb < 2; ++kb) {
    A2p[kb] = load_w2_tile(W2p, kb, lane);
    A2i[kb] = load_w2_tile(W2i, kb, lane);
  }

  // b2p+b2i as initial C of the GEMM2 WMMA chain (C/D layout: state = r + 8*hi)
  v8f cbias;
#pragma unroll
  for (int r = 0; r < 8; ++r) {
    const int s = r + hi * 8;
    cbias[r] = (s < NXS) ? (b2p[s] + b2i[s]) : 0.f;
  }

  // State x in C/D layout: lane<16 holds states 0..7; lane>=16 holds states 8,9 in r=0,1.
  v8f x = {};
  {
    const float* xr = x0 + (size_t)b * NXS;
    if (!hi) {
#pragma unroll
      for (int r = 0; r < 8; ++r) x[r] = xr[r];
    } else {
      x[0] = xr[8];
      x[1] = xr[9];
    }
  }

  const float4* up  = (const float4*)u_b;
  const float4* uip = (const float4*)ui_b;
  const v8f zero = {};
  const unsigned one0 = 0x00003C00u;  // f16 {1.0, 0.0} packed

  for (int t = 0; t < M_STEPS; ++t) {
    const size_t idx = (size_t)t * Q_BATCH + b;
    const float4 u  = up[idx];
    const float4 ui = uip[idx];
    if (t + 1 < M_STEPS) {            // global_prefetch_b8 of next step's inputs
      __builtin_prefetch((const void*)(up + idx + Q_BATCH), 0, 1);
      __builtin_prefetch((const void*)(uip + idx + Q_BATCH), 0, 1);
    }

    // ---- Branchless B operands for GEMM1 (X^T), built as 8 packed-f16 dwords ----
    const unsigned x01 = pk(x[0], x[1]);
    const unsigned x23 = pk(x[2], x[3]);
    const unsigned x45 = pk(x[4], x[5]);
    const unsigned x67 = pk(x[6], x[7]);
    // states 8,9 (f16-packed) fetched from the partner lane-half with ONE shuffle
    const unsigned xs89 = (unsigned)__shfl_xor((int)x01, 16, 32);
    const unsigned u01  = pk(u.x, u.y),  u23  = pk(u.z, u.w);
    const unsigned v01  = pk(ui.x, ui.y), v23 = pk(ui.z, ui.w);

    BU bp;  // pop: K 0..9=x, 10..13=u, 14=1 ; K16..31 = 0 (hi half all zero)
    bp.u[0] = hi ? 0u : x01;  bp.u[1] = hi ? 0u : x23;
    bp.u[2] = hi ? 0u : x45;  bp.u[3] = hi ? 0u : x67;
    bp.u[4] = hi ? 0u : xs89; bp.u[5] = hi ? 0u : u01;
    bp.u[6] = hi ? 0u : u23;  bp.u[7] = hi ? 0u : one0;

    BU bi;  // ind: K 0..9=x, 10..13=u, 14..17=ui, 18=1
    bi.u[0] = hi ? v23  : x01;  // hi: K16,17 = ui2,ui3
    bi.u[1] = hi ? one0 : x23;  // hi: K18 = 1.0
    bi.u[2] = hi ? 0u   : x45;
    bi.u[3] = hi ? 0u   : x67;
    bi.u[4] = hi ? 0u   : xs89;
    bi.u[5] = hi ? 0u   : u01;
    bi.u[6] = hi ? 0u   : u23;
    bi.u[7] = hi ? 0u   : v01;  // lo: K14,15 = ui0,ui1

    // ---- GEMM1: h^T = tanh(W1^T @ X^T + b1) ----
    v8f hp[4], hq[4];
#pragma unroll
    for (int hb = 0; hb < 4; ++hb) hp[hb] = wmma16x16x32(A1p[hb], bp.h, zero);
#pragma unroll
    for (int hb = 0; hb < 4; ++hb) hq[hb] = wmma16x16x32(A1i[hb], bi.h, zero);
#pragma unroll
    for (int hb = 0; hb < 4; ++hb) {
#pragma unroll
      for (int r = 0; r < 8; ++r) {
        hp[hb][r] = tanh_fast(hp[hb][r]);
        hq[hb][r] = tanh_fast(hq[hb][r]);
      }
    }

    // ---- Repack h^T into B-layout (f16), 4 shuffles per tile ----
    const v16h Bp0 = make_bh(hp[0], hp[1], hi);
    const v16h Bp1 = make_bh(hp[2], hp[3], hi);
    const v16h Bi0 = make_bh(hq[0], hq[1], hi);
    const v16h Bi1 = make_bh(hq[2], hq[3], hi);

    // ---- GEMM2 chained: dx^T = b2 + W2p^T@hp^T + W2i^T@hq^T ----
    v8f acc = cbias;
    acc = wmma16x16x32(A2p[0], Bp0, acc);
    acc = wmma16x16x32(A2p[1], Bp1, acc);
    acc = wmma16x16x32(A2i[0], Bi0, acc);
    acc = wmma16x16x32(A2i[1], Bi1, acc);

    // ---- Store pre-update state X_sim[t] (40B row, 8B aligned -> float2 stores) ----
    float* o = out + idx * NXS;
    if (!hi) {
      ((float2*)o)[0] = make_float2(x[0], x[1]);
      ((float2*)o)[1] = make_float2(x[2], x[3]);
      ((float2*)o)[2] = make_float2(x[4], x[5]);
      ((float2*)o)[3] = make_float2(x[6], x[7]);
    } else {
      *((float2*)(o + 8)) = make_float2(x[0], x[1]);
    }

    // ---- Euler update + clamp state 0 (lives in lane<16, r=0) ----
#pragma unroll
    for (int r = 0; r < 8; ++r) x[r] += acc[r];
    const float xc = fminf(fmaxf(x[0], CGM_MIN), CGM_MAX);
    x[0] = hi ? x[0] : xc;
  }
}

extern "C" void kernel_launch(void* const* d_in, const int* in_sizes, int n_in,
                              void* d_out, int out_size, void* d_ws, size_t ws_size,
                              hipStream_t stream) {
  const float* x0  = (const float*)d_in[0];
  const float* u   = (const float*)d_in[1];
  const float* ui  = (const float*)d_in[2];
  const float* W1p = (const float*)d_in[3];
  const float* b1p = (const float*)d_in[4];
  const float* W2p = (const float*)d_in[5];
  const float* b2p = (const float*)d_in[6];
  const float* W1i = (const float*)d_in[7];
  const float* b1i = (const float*)d_in[8];
  const float* W2i = (const float*)d_in[9];
  const float* b2i = (const float*)d_in[10];
  float* out = (float*)d_out;

  // 256 batch tiles of 16 rows; one wave32 per tile, single-wave workgroups
  dim3 grid(Q_BATCH / 16), block(32);
  fes_kernel<<<grid, block, 0, stream>>>(x0, u, ui, W1p, b1p, W2p, b2p,
                                         W1i, b1i, W2i, b2i, out);
}
